// BitwiseTasNetRepeat_70763881169550
// MI455X (gfx1250) — 8-bit matrix operand order repaired and checked by reading, compiled, not run on hardware
//
#include <hip/hip_runtime.h>
#include <stdint.h>

#define EPSV 1e-5f
typedef int v8i __attribute__((ext_vector_type(8)));

__device__ __forceinline__ signed char sign8f(float x) {
    return (signed char)((x > 0.f) - (x < 0.f));
}

__global__ void sign_pack_kernel(const float* __restrict__ in, signed char* __restrict__ out, int n) {
    int i = blockIdx.x * blockDim.x + threadIdx.x;
    if (i < n) out[i] = sign8f(in[i]);
}

__global__ __launch_bounds__(256)
void bn_sign_kernel(const float* __restrict__ x,
                    const float* __restrict__ g, const float* __restrict__ be,
                    const float* __restrict__ m, const float* __restrict__ vv,
                    signed char* __restrict__ out, int C, int T) {
    int bc = blockIdx.x;
    int c  = bc % C;
    float sc = g[c] * rsqrtf(vv[c] + EPSV);
    float sh = be[c] - m[c] * sc;
    const float* row = x + (size_t)bc * T;
    signed char* orow = out + (size_t)bc * T;
    for (int t = threadIdx.x * 4; t < T; t += blockDim.x * 4) {
        float4 f = *(const float4*)(row + t);
        uint32_t w = ((uint32_t)(uint8_t)sign8f(sc * f.x + sh)) |
                     ((uint32_t)(uint8_t)sign8f(sc * f.y + sh) << 8) |
                     ((uint32_t)(uint8_t)sign8f(sc * f.z + sh) << 16) |
                     ((uint32_t)(uint8_t)sign8f(sc * f.w + sh) << 24);
        *(uint32_t*)(orow + t) = w;
    }
}

__global__ __launch_bounds__(256)
void dwconv_bn_sign_kernel(const signed char* __restrict__ xin,
                           const signed char* __restrict__ w,
                           const float* __restrict__ g, const float* __restrict__ be,
                           const float* __restrict__ m, const float* __restrict__ vv,
                           signed char* __restrict__ out, int H, int T, int dil) {
    int bc = blockIdx.x;
    int c  = bc % H;
    float sc = g[c] * rsqrtf(vv[c] + EPSV);
    float sh = be[c] - m[c] * sc;
    int w0 = w[c * 3 + 0], w1 = w[c * 3 + 1], w2 = w[c * 3 + 2];
    const signed char* row = xin + (size_t)bc * T;
    signed char* orow = out + (size_t)bc * T;
    for (int t = threadIdx.x; t < T; t += blockDim.x) {
        int s = (int)row[t] * w1;
        int tl = t - dil; if (tl >= 0) s += (int)row[tl] * w0;
        int tr = t + dil; if (tr < T)  s += (int)row[tr] * w2;
        orow[t] = sign8f(sc * (float)s + sh);
    }
}

#define NTW 64
#define LDS_STRIDE 68
template<int KT, int MODE>
__global__ __launch_bounds__(256)
void gemm_bin_kernel(const signed char* __restrict__ W,
                     const signed char* __restrict__ X,
                     int Mtot, int T,
                     const float* __restrict__ g, const float* __restrict__ be,
                     const float* __restrict__ m, const float* __restrict__ vv,
                     signed char* __restrict__ outs, float* __restrict__ outf) {
    __shared__ int lds_i[NTW * (LDS_STRIDE / 4)];
    char* lds = (char*)lds_i;

    const int tid  = threadIdx.x;
    const int lane = tid & 31;
    const int wv   = tid >> 5;
    const int half = lane >> 4;
    const int l15  = lane & 15;

    const int tiles_per_b = T / NTW;
    const int b  = blockIdx.x / tiles_per_b;
    const int t0 = (blockIdx.x % tiles_per_b) * NTW;
    const int O0 = blockIdx.y * 256 + wv * 32;
    const size_t xbase = (size_t)b * KT * T;

    const int c4 = (tid >> 4) << 2;
    const int t4 = (tid & 15) << 2;

    v8i acc[2][4];
    #pragma unroll
    for (int r = 0; r < 2; ++r)
        #pragma unroll
        for (int f = 0; f < 4; ++f)
            #pragma unroll
            for (int j = 0; j < 8; ++j) acc[r][f][j] = 0;

    for (int k0 = 0; k0 < KT; k0 += 64) {
        __syncthreads();
        {
            const uint8_t* base = (const uint8_t*)X + xbase + (size_t)(k0 + c4) * T + t0 + t4;
            uint32_t r0 = *(const uint32_t*)(base);
            uint32_t r1 = *(const uint32_t*)(base + T);
            uint32_t r2 = *(const uint32_t*)(base + 2 * (size_t)T);
            uint32_t r3 = *(const uint32_t*)(base + 3 * (size_t)T);
            uint32_t lo01 = __builtin_amdgcn_perm(r1, r0, 0x05010400u);
            uint32_t hi01 = __builtin_amdgcn_perm(r1, r0, 0x07030602u);
            uint32_t lo23 = __builtin_amdgcn_perm(r3, r2, 0x05010400u);
            uint32_t hi23 = __builtin_amdgcn_perm(r3, r2, 0x07030602u);
            *(int*)(lds + (t4 + 0) * LDS_STRIDE + c4) = (int)__builtin_amdgcn_perm(lo23, lo01, 0x05040100u);
            *(int*)(lds + (t4 + 1) * LDS_STRIDE + c4) = (int)__builtin_amdgcn_perm(lo23, lo01, 0x07060302u);
            *(int*)(lds + (t4 + 2) * LDS_STRIDE + c4) = (int)__builtin_amdgcn_perm(hi23, hi01, 0x05040100u);
            *(int*)(lds + (t4 + 3) * LDS_STRIDE + c4) = (int)__builtin_amdgcn_perm(hi23, hi01, 0x07060302u);
        }
        __syncthreads();

        v8i a[2];
        #pragma unroll
        for (int r = 0; r < 2; ++r) {
            const signed char* wrow = W + (size_t)(O0 + r * 16 + l15) * KT + k0;
            #pragma unroll
            for (int j = 0; j < 8; ++j) {
                int kb = ((j >> 1) << 4) + ((j & 1) << 2) + (half << 3);
                a[r][j] = *(const int*)(wrow + kb);
            }
        }

        #pragma unroll
        for (int f = 0; f < 4; ++f) {
            v8i bf;
            int nn = f * 16 + l15;
            #pragma unroll
            for (int j = 0; j < 8; ++j) {
                int kb = ((j >> 1) << 4) + ((j & 1) << 2) + (half << 3);
                bf[j] = *(const int*)(lds + nn * LDS_STRIDE + kb);
            }
            acc[0][f] = __builtin_amdgcn_wmma_i32_16x16x64_iu8(
                true, a[0], true, bf, acc[0][f], false, false);
            acc[1][f] = __builtin_amdgcn_wmma_i32_16x16x64_iu8(
                true, a[1], true, bf, acc[1][f], false, false);
        }
    }

    #pragma unroll
    for (int r = 0; r < 2; ++r) {
        #pragma unroll
        for (int f = 0; f < 4; ++f) {
            int t = t0 + f * 16 + l15;
            #pragma unroll
            for (int j = 0; j < 8; ++j) {
                int o = O0 + r * 16 + j + (half << 3);
                if (MODE == 0) {
                    float sc = g[o] * rsqrtf(vv[o] + EPSV);
                    float sh = be[o] - m[o] * sc;
                    outs[(size_t)b * Mtot * T + (size_t)o * T + t] =
                        sign8f(sc * (float)acc[r][f][j] + sh);
                } else {
                    size_t idx = (size_t)b * Mtot * T + (size_t)o * T + t;
                    outf[idx] += (float)acc[r][f][j];
                }
            }
        }
    }
}

extern "C" void kernel_launch(void* const* d_in, const int* in_sizes, int n_in,
                              void* d_out, int out_size, void* d_ws, size_t ws_size,
                              hipStream_t stream) {
    const int B = 16, Cb = 256, H = 512, T = 4096, NBLK = 8;

    const float* x    = (const float*)d_in[0];
    const float* bn1g = (const float*)d_in[1];
    const float* bn1b = (const float*)d_in[2];
    const float* bn1m = (const float*)d_in[3];
    const float* bn1v = (const float*)d_in[4];
    const float* w1   = (const float*)d_in[5];
    const float* bn2g = (const float*)d_in[6];
    const float* bn2b = (const float*)d_in[7];
    const float* bn2m = (const float*)d_in[8];
    const float* bn2v = (const float*)d_in[9];
    const float* dww  = (const float*)d_in[10];
    const float* bn3g = (const float*)d_in[11];
    const float* bn3b = (const float*)d_in[12];
    const float* bn3m = (const float*)d_in[13];
    const float* bn3v = (const float*)d_in[14];
    const float* w2   = (const float*)d_in[15];

    float* resid = (float*)d_out;

    char*  ws  = (char*)d_ws;
    size_t off = 0;
    signed char* sw1 = (signed char*)(ws + off); off += (size_t)NBLK * H * Cb;
    signed char* sw2 = (signed char*)(ws + off); off += (size_t)NBLK * Cb * H;
    signed char* sdw = (signed char*)(ws + off); off += 16384;
    signed char* s1  = (signed char*)(ws + off); off += (size_t)B * Cb * T;
    signed char* s2  = (signed char*)(ws + off); off += (size_t)B * H * T;
    signed char* s3  = (signed char*)(ws + off); off += (size_t)B * H * T;

    hipMemcpyAsync(d_out, x, (size_t)B * Cb * T * sizeof(float),
                   hipMemcpyDeviceToDevice, stream);

    { int n = NBLK * H * Cb;
      sign_pack_kernel<<<(n + 255) / 256, 256, 0, stream>>>(w1, sw1, n); }
    { int n = NBLK * Cb * H;
      sign_pack_kernel<<<(n + 255) / 256, 256, 0, stream>>>(w2, sw2, n); }
    { int n = NBLK * H * 3;
      sign_pack_kernel<<<(n + 255) / 256, 256, 0, stream>>>(dww, sdw, n); }

    for (int i = 0; i < NBLK; ++i) {
        bn_sign_kernel<<<B * Cb, 256, 0, stream>>>(
            resid, bn1g + i * Cb, bn1b + i * Cb, bn1m + i * Cb, bn1v + i * Cb,
            s1, Cb, T);

        dim3 g1(B * (T / NTW), H / 256);
        gemm_bin_kernel<Cb, 0><<<g1, 256, 0, stream>>>(
            sw1 + (size_t)i * H * Cb, s1, H, T,
            bn2g + i * H, bn2b + i * H, bn2m + i * H, bn2v + i * H,
            s2, nullptr);

        dwconv_bn_sign_kernel<<<B * H, 256, 0, stream>>>(
            s2, sdw + (size_t)i * H * 3,
            bn3g + i * H, bn3b + i * H, bn3m + i * H, bn3v + i * H,
            s3, H, T, 1 << i);

        dim3 g2(B * (T / NTW), Cb / 256);
        gemm_bin_kernel<H, 1><<<g2, 256, 0, stream>>>(
            sw2 + (size_t)i * Cb * H, s3, Cb, T,
            nullptr, nullptr, nullptr, nullptr,
            nullptr, resid);
    }
}
